// TrafficPredictorGNN_26431228739926
// MI455X (gfx1250) — compile-verified, matrix operand order checked by reading, not run on hardware
//
#include <hip/hip_runtime.h>
#include <hip/hip_bf16.h>

#define HF 128

typedef __attribute__((ext_vector_type(16))) __bf16 v16bf;
typedef __attribute__((ext_vector_type(8)))  float  v8f;

__device__ __forceinline__ unsigned short f2bf(float f) {
    unsigned u = __float_as_uint(f);
    unsigned r = u + 0x7fffu + ((u >> 16) & 1u);   // RNE
    return (unsigned short)(r >> 16);
}

// order-preserving f32 -> u32 map for atomic max
__device__ __forceinline__ unsigned enc_f(float f) {
    unsigned u = __float_as_uint(f);
    return u ^ ((unsigned)((int)u >> 31) | 0x80000000u);
}

// ---------------- zero workspace region ----------------
__global__ void zero_u4(uint4* p, int n4) {
    int t = blockIdx.x * blockDim.x + threadIdx.x;
    if (t < n4) p[t] = make_uint4(0u, 0u, 0u, 0u);
}

// ---------------- edge scatter-max (wave32 per edge) ----------------
__global__ void scatter_max_kernel(const float* __restrict__ h,
                                   const int* __restrict__ ei,
                                   unsigned* __restrict__ m_enc, int E) {
    int t = blockIdx.x * blockDim.x + threadIdx.x;
    int e = t >> 5;
    int lane = t & 31;
    if (e >= E) return;
    int src = ei[e];
    int dst = ei[E + e];
    const float4 v = *(const float4*)(h + (size_t)src * HF + lane * 4);
    unsigned* d = m_enc + (size_t)dst * HF + lane * 4;
    atomicMax(d + 0, enc_f(v.x));
    atomicMax(d + 1, enc_f(v.y));
    atomicMax(d + 2, enc_f(v.z));
    atomicMax(d + 3, enc_f(v.w));
}

// ---------------- pack 128x128 f32 weight into B-fragment bf16 order ----------------
// Bpacked[((t*4+s)*32+L)*16 + r*2 + j] = bf16( W[o*128+k] )
//   o = t*16 + (L&15),  k = s*32 + 16*(r>=4) + 8*(L>=16) + 2*(r&3) + j
__global__ void pack_weight_kernel(const float* __restrict__ W,
                                   unsigned short* __restrict__ out) {
    int p = blockIdx.x * blockDim.x + threadIdx.x;   // 0..16383
    if (p >= 8 * 4 * 32 * 16) return;
    int j = p & 1;
    int r = (p >> 1) & 7;
    int L = (p >> 4) & 31;
    int s = (p >> 9) & 3;
    int tt = p >> 11;
    int o = tt * 16 + (L & 15);
    int kk = ((r >> 2) << 4) + ((L >> 4) << 3) + ((r & 3) << 1) + j;
    int k = s * 32 + kk;
    out[p] = f2bf(W[o * HF + k]);
}

// ---------------- fused SAGE layer GEMM ----------------
// out = elu( decode(m_enc) @ Wl^T + b + h_in @ Wr^T ) [+ resid]
// block = 256 threads = 8 waves; one 16-node x 128-out tile per block
__global__ void sage_gemm_kernel(const unsigned* __restrict__ m_enc,
                                 const float* __restrict__ h_in,
                                 const unsigned short* __restrict__ Bl,
                                 const unsigned short* __restrict__ Br,
                                 const float* __restrict__ bias,
                                 const float* __restrict__ resid,   // may be null
                                 float* __restrict__ h_out, int nodes) {
    __shared__ __align__(32) unsigned short lds_m[4 * 32 * 16];  // 4 KB, A-frag order
    __shared__ __align__(32) unsigned short lds_h[4 * 32 * 16];  // 4 KB

    const int node0 = blockIdx.x * 16;
    const int tid = threadIdx.x;

    // cooperative stage of 16x128 A-tiles (both matrices), pre-swizzled to fragment layout
    for (int i = 0; i < 8; ++i) {
        int idx = tid + i * 256;          // 0..2047
        int M = idx >> 7;                 // node row in tile
        int k = idx & 127;
        int s = k >> 5;
        int kk = k & 31;
        int r = (((kk >> 4) & 1) << 2) + ((kk >> 1) & 3);
        int L = (((kk >> 3) & 1) << 4) + M;
        int j = kk & 1;
        int fo = ((s * 32 + L) * 8 + r) * 2 + j;
        int node = node0 + M;
        if (node >= nodes) node = nodes - 1;
        size_t g = (size_t)node * HF + k;

        unsigned u = m_enc[g];
        unsigned b = (u >> 31) ? (u ^ 0x80000000u) : ~u;  // decode order-map
        float fm = __uint_as_float(b);
        if ((b & 0x7f800000u) == 0x7f800000u) fm = 0.0f;  // empty segment / non-finite -> 0
        lds_m[fo] = f2bf(fm);
        lds_h[fo] = f2bf(h_in[g]);
    }
    __syncthreads();

    const int wave = tid >> 5;   // output-column tile 0..7
    const int lane = tid & 31;

    v8f c = {};
    #pragma unroll
    for (int s = 0; s < 4; ++s) {
        v16bf am = *(const v16bf*)(lds_m + (s * 32 + lane) * 16);
        v16bf bl = *(const v16bf*)(Bl + ((size_t)(wave * 4 + s) * 32 + lane) * 16);
        c = __builtin_amdgcn_wmma_f32_16x16x32_bf16(false, am, false, bl, (short)0, c, false, false);
        v16bf ah = *(const v16bf*)(lds_h + (s * 32 + lane) * 16);
        v16bf br = *(const v16bf*)(Br + ((size_t)(wave * 4 + s) * 32 + lane) * 16);
        c = __builtin_amdgcn_wmma_f32_16x16x32_bf16(false, ah, false, br, (short)0, c, false, false);
    }

    const int o = wave * 16 + (lane & 15);
    const float bo = bias[o];
    const int mbase = (lane >> 4) * 8;   // lanes 0-15: M=r ; lanes 16-31: M=r+8
    #pragma unroll
    for (int r = 0; r < 8; ++r) {
        int node = node0 + mbase + r;
        if (node >= nodes) continue;
        float v = c[r] + bo;
        v = v > 0.0f ? v : (__expf(v) - 1.0f);          // ELU
        if (resid) v += resid[(size_t)node * HF + o];   // residual
        h_out[(size_t)node * HF + o] = v;
    }
}

// ---------------- final 128->1 GEMV + sigmoid (wave32 per node) ----------------
__global__ void out_head_kernel(const float* __restrict__ h3,
                                const float* __restrict__ Wo,
                                const float* __restrict__ bo,
                                float* __restrict__ out, int nodes) {
    int t = blockIdx.x * blockDim.x + threadIdx.x;
    int node = t >> 5;
    int lane = t & 31;
    if (node >= nodes) return;
    float4 a = *(const float4*)(h3 + (size_t)node * HF + lane * 4);
    float4 w = *(const float4*)(Wo + lane * 4);
    float d = a.x * w.x + a.y * w.y + a.z * w.z + a.w * w.w;
    #pragma unroll
    for (int off = 16; off > 0; off >>= 1) d += __shfl_xor(d, off, 32);
    if (lane == 0) out[node] = 1.0f / (1.0f + __expf(-(d + bo[0])));
}

extern "C" void kernel_launch(void* const* d_in, const int* in_sizes, int n_in,
                              void* d_out, int out_size, void* d_ws, size_t ws_size,
                              hipStream_t stream) {
    const float* x   = (const float*)d_in[0];
    const int*   ei  = (const int*)d_in[1];
    const float* W1l = (const float*)d_in[2];
    const float* b1  = (const float*)d_in[3];
    const float* W1r = (const float*)d_in[4];
    const float* W2l = (const float*)d_in[5];
    const float* b2  = (const float*)d_in[6];
    const float* W2r = (const float*)d_in[7];
    const float* W3l = (const float*)d_in[8];
    const float* b3  = (const float*)d_in[9];
    const float* W3r = (const float*)d_in[10];
    const float* Wo  = (const float*)d_in[11];
    const float* bo  = (const float*)d_in[12];

    const int N = in_sizes[0] / HF;
    const int E = in_sizes[1] / 2;

    // workspace layout
    char* base = (char*)d_ws;
    size_t featB = (size_t)N * HF * sizeof(float);
    unsigned* m_enc = (unsigned*)base;                 base += (featB + 255) & ~255ull;
    float* bufA     = (float*)base;                    base += (featB + 255) & ~255ull;
    float* bufB     = (float*)base;                    base += (featB + 255) & ~255ull;
    const size_t packB = 8 * 4 * 32 * 16 * sizeof(unsigned short); // 32 KB
    unsigned short* P[6];
    for (int i = 0; i < 6; ++i) { P[i] = (unsigned short*)base; base += packB; }

    // pack the six weight matrices into B-fragment bf16 order
    const float* Ws[6] = {W1l, W1r, W2l, W2r, W3l, W3r};
    for (int i = 0; i < 6; ++i)
        pack_weight_kernel<<<64, 256, 0, stream>>>(Ws[i], P[i]);

    const int feat4 = N * HF / 4;
    const int zgrid = (feat4 + 255) / 256;
    const int egrid = (int)(((size_t)E * 32 + 255) / 256);
    const int tgrid = (N + 15) / 16;

    // ---- layer 1: x -> bufA (no residual)
    zero_u4<<<zgrid, 256, 0, stream>>>((uint4*)m_enc, feat4);
    scatter_max_kernel<<<egrid, 256, 0, stream>>>(x, ei, m_enc, E);
    sage_gemm_kernel<<<tgrid, 256, 0, stream>>>(m_enc, x, P[0], P[1], b1, nullptr, bufA, N);

    // ---- layer 2: bufA -> bufB (residual = bufA)
    zero_u4<<<zgrid, 256, 0, stream>>>((uint4*)m_enc, feat4);
    scatter_max_kernel<<<egrid, 256, 0, stream>>>(bufA, ei, m_enc, E);
    sage_gemm_kernel<<<tgrid, 256, 0, stream>>>(m_enc, bufA, P[2], P[3], b2, bufA, bufB, N);

    // ---- layer 3: bufB -> bufA (residual = bufB)
    zero_u4<<<zgrid, 256, 0, stream>>>((uint4*)m_enc, feat4);
    scatter_max_kernel<<<egrid, 256, 0, stream>>>(bufB, ei, m_enc, E);
    sage_gemm_kernel<<<tgrid, 256, 0, stream>>>(m_enc, bufB, P[4], P[5], b3, bufB, bufA, N);

    // ---- output head
    const int ogrid = (int)(((size_t)N * 32 + 255) / 256);
    out_head_kernel<<<ogrid, 256, 0, stream>>>(bufA, Wo, bo, (float*)d_out, N);
}
